// SelfAttention_2886218023621
// MI455X (gfx1250) — matrix operand order repaired and checked by reading, not compiled, not run on hardware
//
#include <hip/hip_runtime.h>

#define EMBED   512
#define DHEAD   64
#define BATCH   2
#define SEQ     4096
#define NROWS   (BATCH * SEQ)      // 8192 total rows
#define NTILES  (NROWS / 16)       // 512 query tiles
#define SEG     4
#define KEYS_PER_SEG (SEQ / SEG)   // 1024
#define NCHUNK  (KEYS_PER_SEG / 32) // 32 chunks of 32 keys

typedef __attribute__((ext_vector_type(16))) __bf16 v16bf;
typedef __attribute__((ext_vector_type(8)))  __bf16 v8bf;
typedef __attribute__((ext_vector_type(8)))  float  v8f;
typedef __attribute__((ext_vector_type(4)))  float  v4f;
typedef __attribute__((ext_vector_type(4)))  unsigned int u32x4;
typedef __attribute__((ext_vector_type(8)))  int    i32x8;
typedef __attribute__((ext_vector_type(4)))  int    i32x4;

__device__ __forceinline__ v8f wmma_bf16(v16bf a, v16bf b, v8f c) {
  // D = A(16x32 bf16) * B(32x16 bf16) + C(16x16 f32)
  return __builtin_amdgcn_wmma_f32_16x16x32_bf16(
      /*neg_a=*/false, a, /*neg_b=*/false, b,
      /*c_mod=*/(short)0, c, /*reuse_a=*/false, /*reuse_b=*/false);
}

// A-operand (16x32 bf16) gather from a row-major [*,64] bf16 matrix.
__device__ __forceinline__ v16bf load_a_bf16(const __bf16* rowptr, int hi, int d0) {
  v8bf g0 = *(const v8bf*)(rowptr + d0 + 16 * hi);
  v8bf g1 = *(const v8bf*)(rowptr + d0 + 16 * hi + 8);
  v16bf a;
#pragma unroll
  for (int i = 0; i < 8; ++i) { a[i] = g0[i]; a[i + 8] = g1[i]; }
  return a;
}

// ---------------------------------------------------------------------------
// TDM: DMA a 2D bf16 tile (tile_d1 rows x tile_d0 elems) from a row-major
// tensor (row stride = stride0 elems) into LDS at lds_addr.  D# bit layout
// per CDNA5 ISA (data_size=1 -> 2 bytes).
// ---------------------------------------------------------------------------
__device__ __forceinline__ void tdm_load_2d_bf16(
    unsigned lds_addr, const void* gptr,
    unsigned tensor_d0, unsigned tensor_d1,
    unsigned tile_d0, unsigned tile_d1, unsigned stride0)
{
  unsigned long long ga = (unsigned long long)gptr;
  u32x4 g0 = {
    1u,                                              // count=1, user descriptor
    lds_addr,                                        // [63:32]  LDS byte address
    (unsigned)(ga & 0xFFFFFFFFu),                    // [95:64]  global addr lo
    (unsigned)((ga >> 32) & 0x1FFFFFFu) | (2u << 30) // [120:96] addr hi, type=2
  };
  i32x8 g1 = {
    (int)(1u << 16),                                              // data_size=1 (2B)
    (int)((tensor_d0 & 0xFFFFu) << 16),                           // tensor_dim0 lo
    (int)(((tensor_d0 >> 16) & 0xFFFFu) | ((tensor_d1 & 0xFFFFu) << 16)),
    (int)(((tensor_d1 >> 16) & 0xFFFFu) | ((tile_d0 & 0xFFFFu) << 16)),
    (int)(tile_d1 & 0xFFFFu),                                     // tile_dim1, tile_dim2=0
    (int)stride0,                                                 // dim0 stride lo
    0, 0                                                          // stride hi, dim1 stride
  };
  i32x4 gz = {0, 0, 0, 0};
#if defined(__clang_major__) && (__clang_major__ >= 23)
  i32x8 gz8 = {0, 0, 0, 0, 0, 0, 0, 0};
  __builtin_amdgcn_tensor_load_to_lds(g0, g1, gz, gz, gz8, 0);
#else
  __builtin_amdgcn_tensor_load_to_lds(g0, g1, gz, gz, 0);
#endif
}

// ---------------------------------------------------------------------------
// Kernel 1: QKV projection.  [8192,512] fp32 @ [512,64] -> bf16.
// grid = (512 row-tiles, 3 matrices), block = 128 (4 waves, one 16-col tile each)
// Q,K stored row-major [8192,64]; V stored transposed [B,64,S].
// ---------------------------------------------------------------------------
__global__ __launch_bounds__(128) void qkv_proj_kernel(
    const float* __restrict__ X,  const float* __restrict__ Wq,
    const float* __restrict__ Wk, const float* __restrict__ Wv,
    __bf16* __restrict__ Qb, __bf16* __restrict__ Kb, __bf16* __restrict__ Vt)
{
  const int wave = threadIdx.x >> 5;
  const int lane = threadIdx.x & 31;
  const int ln   = lane & 15;
  const int hi   = lane >> 4;
  const int row0 = blockIdx.x * 16;
  const int mat  = blockIdx.y;          // 0=Q 1=K 2=V
  const int n0   = wave * 16;           // output-unit tile

  const float* W = (mat == 0) ? Wq : (mat == 1) ? Wk : Wv;
  const float* xrow = X + (size_t)(row0 + ln) * EMBED;
  const float* wrow = W + (size_t)(n0 + ln) * EMBED;

  v8f acc = {};
#pragma unroll 4
  for (int e0 = 0; e0 < EMBED; e0 += 32) {
    v4f a0 = *(const v4f*)(xrow + e0 + 16 * hi);
    v4f a1 = *(const v4f*)(xrow + e0 + 16 * hi + 4);
    v4f a2 = *(const v4f*)(xrow + e0 + 16 * hi + 8);
    v4f a3 = *(const v4f*)(xrow + e0 + 16 * hi + 12);
    v16bf a;
#pragma unroll
    for (int i = 0; i < 4; ++i) {
      a[i]      = (__bf16)a0[i];
      a[i + 4]  = (__bf16)a1[i];
      a[i + 8]  = (__bf16)a2[i];
      a[i + 12] = (__bf16)a3[i];
    }
    v4f b0 = *(const v4f*)(wrow + e0 + 16 * hi);
    v4f b1 = *(const v4f*)(wrow + e0 + 16 * hi + 4);
    v4f b2 = *(const v4f*)(wrow + e0 + 16 * hi + 8);
    v4f b3 = *(const v4f*)(wrow + e0 + 16 * hi + 12);
    v16bf b;
#pragma unroll
    for (int i = 0; i < 4; ++i) {
      b[i]      = (__bf16)b0[i];
      b[i + 4]  = (__bf16)b1[i];
      b[i + 8]  = (__bf16)b2[i];
      b[i + 12] = (__bf16)b3[i];
    }
    acc = wmma_bf16(a, b, acc);
  }

  if (mat < 2) {
    __bf16* obuf = (mat == 0) ? Qb : Kb;
#pragma unroll
    for (int r = 0; r < 8; ++r)
      obuf[(size_t)(row0 + r + 8 * hi) * DHEAD + n0 + ln] = (__bf16)acc[r];
  } else {
#pragma unroll
    for (int r = 0; r < 8; ++r) {
      int sg = row0 + r + 8 * hi;
      int bb = sg >> 12;
      int ss = sg & (SEQ - 1);
      Vt[((size_t)bb * DHEAD + (n0 + ln)) * SEQ + ss] = (__bf16)acc[r];
    }
  }
}

// ---------------------------------------------------------------------------
// Kernel 2: flash attention partials.  block = 128 (4 waves), each wave owns
// one 16-query tile; all 4 waves share the SAME key segment, so K/V chunks
// are staged once per block into LDS by the Tensor Data Mover (wave 0 issues
// the descriptors), double-buffered, synced with s_wait_tensorcnt + barriers.
// grid = (NTILES/4, SEG).
// ---------------------------------------------------------------------------
__global__ __launch_bounds__(128) void flash_attn_kernel(
    const __bf16* __restrict__ Qb, const __bf16* __restrict__ Kb,
    const __bf16* __restrict__ Vt,
    float* __restrict__ Opart, float* __restrict__ Mpart,
    float* __restrict__ Lpart)
{
  __shared__ __attribute__((aligned(64))) __bf16 Kbuf[2][32 * DHEAD]; // [key][d]
  __shared__ __attribute__((aligned(64))) __bf16 Vbuf[2][DHEAD * 32]; // [d][key]
  __shared__ __attribute__((aligned(32))) __bf16 Pbuf[4][16 * 40];    // per-wave P

  const int wave = threadIdx.x >> 5;
  const int lane = threadIdx.x & 31;
  const int ln   = lane & 15;
  const int hi   = lane >> 4;
  const int qt   = blockIdx.x * 4 + wave;        // 0..511 (same batch per block)
  const int seg  = blockIdx.y;
  const int q0g  = qt * 16;
  const int bb   = q0g >> 12;
  const int krow0 = bb * SEQ + seg * KEYS_PER_SEG;
  const __bf16* kglob = Kb + (size_t)krow0 * DHEAD;
  const __bf16* vglob = Vt + (size_t)bb * DHEAD * SEQ + seg * KEYS_PER_SEG;

  const unsigned lds_k0 = (unsigned)(size_t)&Kbuf[0][0];
  const unsigned lds_k1 = (unsigned)(size_t)&Kbuf[1][0];
  const unsigned lds_v0 = (unsigned)(size_t)&Vbuf[0][0];
  const unsigned lds_v1 = (unsigned)(size_t)&Vbuf[1][0];

  // Prologue: wave 0 DMAs chunks 0 and 1 (K: 32x64 tile, V: 64x32 tile).
  if (wave == 0) {
    tdm_load_2d_bf16(lds_k0, kglob,              DHEAD, SEQ,  DHEAD, 32, DHEAD);
    tdm_load_2d_bf16(lds_v0, vglob,              SEQ,   DHEAD, 32,   DHEAD, SEQ);
    tdm_load_2d_bf16(lds_k1, kglob + 32 * DHEAD, DHEAD, SEQ,  DHEAD, 32, DHEAD);
    tdm_load_2d_bf16(lds_v1, vglob + 32,         SEQ,   DHEAD, 32,   DHEAD, SEQ);
  }

  // Q A-operands: d 0..31 and 32..63 (resident for the whole loop)
  const __bf16* qrow = Qb + (size_t)(q0g + ln) * DHEAD;
  const v16bf qa0 = load_a_bf16(qrow, hi, 0);
  const v16bf qa1 = load_a_bf16(qrow, hi, 32);

  float m[8], l[8];
#pragma unroll
  for (int r = 0; r < 8; ++r) { m[r] = -1e30f; l[r] = 0.0f; }
  v8f o0 = {}, o1 = {}, o2 = {}, o3 = {};

  __bf16* pb = &Pbuf[wave][0];

  for (int i = 0; i < NCHUNK; ++i) {
    const int cur = i & 1;
    if (wave == 0) {
      if (i + 1 < NCHUNK) __builtin_amdgcn_s_wait_tensorcnt(2); // chunk i landed
      else                __builtin_amdgcn_s_wait_tensorcnt(0); // last chunk
    }
    __syncthreads();

    const __bf16* lk = &Kbuf[cur][0];
    const __bf16* lv = &Vbuf[cur][0];

    // ---- scores S = (Q K^T) * 8 for this 16x32 key chunk (from LDS) ------
    v16bf kb00 = *(const v16bf*)(lk + (size_t)ln        * DHEAD + 16 * hi);
    v16bf kb01 = *(const v16bf*)(lk + (size_t)ln        * DHEAD + 32 + 16 * hi);
    v16bf kb10 = *(const v16bf*)(lk + (size_t)(16 + ln) * DHEAD + 16 * hi);
    v16bf kb11 = *(const v16bf*)(lk + (size_t)(16 + ln) * DHEAD + 32 + 16 * hi);

    v8f s0 = {}, s1 = {};
    s0 = wmma_bf16(qa0, kb00, s0);
    s0 = wmma_bf16(qa1, kb01, s0);
    s1 = wmma_bf16(qa0, kb10, s1);
    s1 = wmma_bf16(qa1, kb11, s1);
    s0 = s0 * 8.0f;                 // faithful quirk: scores / (64**-0.5)
    s1 = s1 * 8.0f;

    // ---- online softmax (row stats across 16-lane groups) ----------------
    float alpha[8];
#pragma unroll
    for (int r = 0; r < 8; ++r) {
      float cm = fmaxf(s0[r], s1[r]);
#pragma unroll
      for (int off = 1; off < 16; off <<= 1)
        cm = fmaxf(cm, __shfl_xor(cm, off, 32));
      float mn = fmaxf(m[r], cm);
      alpha[r] = __expf(m[r] - mn);
      float p0 = __expf(s0[r] - mn);
      float p1 = __expf(s1[r] - mn);
      float rs = p0 + p1;
#pragma unroll
      for (int off = 1; off < 16; off <<= 1)
        rs += __shfl_xor(rs, off, 32);
      l[r] = l[r] * alpha[r] + rs;
      m[r] = mn;
      int prow = r + 8 * hi;
      pb[prow * 40 + ln]      = (__bf16)p0;
      pb[prow * 40 + 16 + ln] = (__bf16)p1;
    }

    // ---- reload P in A-matrix layout (per-wave LDS, no barrier needed) ---
    const __bf16* pr = pb + ln * 40;
    v8bf g0 = *(const v8bf*)(pr + 16 * hi);
    v8bf g1 = *(const v8bf*)(pr + 16 * hi + 8);
    v16bf pa;
#pragma unroll
    for (int k = 0; k < 8; ++k) { pa[k] = g0[k]; pa[k + 8] = g1[k]; }

    // ---- rescale O, accumulate P @ V (V tile from LDS) -------------------
#pragma unroll
    for (int r = 0; r < 8; ++r) {
      o0[r] *= alpha[r]; o1[r] *= alpha[r];
      o2[r] *= alpha[r]; o3[r] *= alpha[r];
    }
    v16bf vb0 = *(const v16bf*)(lv + (size_t)(ln)      * 32 + 16 * hi);
    v16bf vb1 = *(const v16bf*)(lv + (size_t)(16 + ln) * 32 + 16 * hi);
    v16bf vb2 = *(const v16bf*)(lv + (size_t)(32 + ln) * 32 + 16 * hi);
    v16bf vb3 = *(const v16bf*)(lv + (size_t)(48 + ln) * 32 + 16 * hi);
    o0 = wmma_bf16(pa, vb0, o0);
    o1 = wmma_bf16(pa, vb1, o1);
    o2 = wmma_bf16(pa, vb2, o2);
    o3 = wmma_bf16(pa, vb3, o3);

    __syncthreads();   // everyone done with buf[cur] before it is re-filled
    if (wave == 0 && i + 2 < NCHUNK) {
      const int kc2 = (i + 2) * 32;
      tdm_load_2d_bf16(cur ? lds_k1 : lds_k0, kglob + (size_t)kc2 * DHEAD,
                       DHEAD, SEQ, DHEAD, 32, DHEAD);
      tdm_load_2d_bf16(cur ? lds_v1 : lds_v0, vglob + kc2,
                       SEQ, DHEAD, 32, DHEAD, SEQ);
    }
  }

  // ---- write segment partials --------------------------------------------
  const int pbase = (qt * SEG + seg) * 16;
#pragma unroll
  for (int r = 0; r < 8; ++r) {
    int row = r + 8 * hi;
    float* op = Opart + (size_t)(pbase + row) * DHEAD;
    op[ 0 + ln] = o0[r];
    op[16 + ln] = o1[r];
    op[32 + ln] = o2[r];
    op[48 + ln] = o3[r];
    Mpart[pbase + row] = m[r];   // 16 lanes store identical value: benign
    Lpart[pbase + row] = l[r];
  }
}

// ---------------------------------------------------------------------------
// Kernel 3: merge SEG partials with log-sum-exp correction, normalize.
// ---------------------------------------------------------------------------
__global__ __launch_bounds__(256) void attn_reduce_kernel(
    const float* __restrict__ Opart, const float* __restrict__ Mpart,
    const float* __restrict__ Lpart, float* __restrict__ Out)
{
  const int qt  = blockIdx.x;
  const int row = threadIdx.x >> 4;
  const int cg  = threadIdx.x & 15;
  const int pbase = qt * SEG * 16;

  float mg = -1e30f;
#pragma unroll
  for (int s = 0; s < SEG; ++s)
    mg = fmaxf(mg, Mpart[pbase + s * 16 + row]);

  float lsum = 0.0f;
  float acc[4] = {0.f, 0.f, 0.f, 0.f};
#pragma unroll
  for (int s = 0; s < SEG; ++s) {
    float w = __expf(Mpart[pbase + s * 16 + row] - mg);
    lsum += Lpart[pbase + s * 16 + row] * w;
    v4f v = *(const v4f*)(Opart + (size_t)(pbase + s * 16 + row) * DHEAD + cg * 4);
#pragma unroll
    for (int i = 0; i < 4; ++i) acc[i] += v[i] * w;
  }
  float inv = 1.0f / lsum;
  v4f o;
#pragma unroll
  for (int i = 0; i < 4; ++i) o[i] = acc[i] * inv;
  *(v4f*)(Out + (size_t)(qt * 16 + row) * DHEAD + cg * 4) = o;
}

// ---------------------------------------------------------------------------
extern "C" void kernel_launch(void* const* d_in, const int* in_sizes, int n_in,
                              void* d_out, int out_size, void* d_ws, size_t ws_size,
                              hipStream_t stream)
{
  const float* X  = (const float*)d_in[0];
  const float* Wq = (const float*)d_in[1];
  const float* Wk = (const float*)d_in[2];
  const float* Wv = (const float*)d_in[3];

  char* ws = (char*)d_ws;
  __bf16* Qb = (__bf16*)ws;  ws += (size_t)NROWS * DHEAD * sizeof(__bf16);
  __bf16* Kb = (__bf16*)ws;  ws += (size_t)NROWS * DHEAD * sizeof(__bf16);
  __bf16* Vt = (__bf16*)ws;  ws += (size_t)NROWS * DHEAD * sizeof(__bf16);
  float* Opart = (float*)ws; ws += (size_t)NTILES * SEG * 16 * DHEAD * sizeof(float);
  float* Mpart = (float*)ws; ws += (size_t)NTILES * SEG * 16 * sizeof(float);
  float* Lpart = (float*)ws;

  qkv_proj_kernel<<<dim3(NROWS / 16, 3), 128, 0, stream>>>(X, Wq, Wk, Wv, Qb, Kb, Vt);
  flash_attn_kernel<<<dim3(NTILES / 4, SEG), 128, 0, stream>>>(Qb, Kb, Vt, Opart, Mpart, Lpart);
  attn_reduce_kernel<<<NTILES, 256, 0, stream>>>(Opart, Mpart, Lpart, (float*)d_out);
}
